// MultiheadAttention_49864570306861
// MI455X (gfx1250) — matrix operand order repaired and checked by reading, not compiled, not run on hardware
//
#include <hip/hip_runtime.h>

// ---------------------------------------------------------------------------
// MI455X (gfx1250): multi-head-attention-over-heads.
//   - one-pass fp32 -> bf16 conversion of activations/weights
//   - bf16 WMMA GEMMs with double-buffered ASYNC global->LDS tile copies
//   - exact fp32 per-token 16x16 softmax attention over the heads axis
// ---------------------------------------------------------------------------

typedef __bf16 bf16_t;
typedef bf16_t v16bf __attribute__((ext_vector_type(16)));
typedef float  v8f   __attribute__((ext_vector_type(8)));

#define TILE_M 128
#define TILE_N 128
#define TILE_K 32
#define LDSN   40   // ushorts per LDS row: 32 data + 8 pad (80B rows -> conflict-free b128)

// pointer types for the async-copy builtin: (global int4*, local int4*, imm, imm)
typedef int v4i __attribute__((__vector_size__(16)));
typedef __attribute__((address_space(1))) v4i as1_v4i;
typedef __attribute__((address_space(3))) v4i as3_v4i;

// fp32 -> bf16 RNE
__device__ __forceinline__ unsigned short f2bf(float f) {
    union { float f; unsigned u; } v; v.f = f;
    return (unsigned short)((v.u + 0x7FFFu + ((v.u >> 16) & 1u)) >> 16);
}
__device__ __forceinline__ unsigned f2bf_pack(float lo, float hi) {
    return (unsigned)f2bf(lo) | ((unsigned)f2bf(hi) << 16);
}

// ---- CDNA5 async global->LDS copy (ASYNCcnt-tracked), with asm fallback ----
__device__ __forceinline__ void async_copy_b128(const void* g, const void* l) {
#if __has_builtin(__builtin_amdgcn_global_load_async_to_lds_b128)
    __builtin_amdgcn_global_load_async_to_lds_b128(
        (as1_v4i*)(unsigned long long)g,
        (as3_v4i*)(unsigned)(unsigned long long)l, 0, 0);
#else
    asm volatile("global_load_async_to_lds_b128 %0, %1, off"
                 :: "v"((unsigned)(unsigned long long)l),
                    "v"((unsigned long long)g)
                 : "memory");
#endif
}
__device__ __forceinline__ void wait_async0() {
#if __has_builtin(__builtin_amdgcn_s_wait_asynccnt)
    __builtin_amdgcn_s_wait_asynccnt(0);
#else
    asm volatile("s_wait_asynccnt 0x0" ::: "memory");
#endif
}

union Frag { uint4 u[2]; v16bf v; };

// ---------------------------------------------------------------------------
// elementwise fp32 -> packed bf16 (4 floats / thread)
// ---------------------------------------------------------------------------
__global__ __launch_bounds__(256)
void convert_f32_bf16(const float* __restrict__ src, unsigned* __restrict__ dst,
                      long nf4)
{
    long i = (long)blockIdx.x * blockDim.x + threadIdx.x;
    if (i < nf4) {
        float4 v = ((const float4*)src)[i];
        uint2 p;
        p.x = f2bf_pack(v.x, v.y);
        p.y = f2bf_pack(v.z, v.w);
        ((uint2*)dst)[i] = p;
    }
}

// ---------------------------------------------------------------------------
// C[M,N] = A[M,K](bf16) @ W[N,K](bf16)^T + bias, fp32 out.
// 256 threads = 8 wave32; block tile 128x128x32; wave tile 32x64 (2x4 WMMA).
// Double-buffered async LDS tiles: copy engine fills buf^1 during WMMA on buf.
// ---------------------------------------------------------------------------
__global__ __launch_bounds__(256)
void gemm_bf16_wmma(const unsigned short* __restrict__ A,
                    const unsigned short* __restrict__ W,
                    const float* __restrict__ bias,
                    float* __restrict__ C,
                    int M, int N, int K)
{
    __shared__ unsigned short As[2][TILE_M * LDSN];
    __shared__ unsigned short Bs[2][TILE_N * LDSN];

    const int tid  = threadIdx.x;
    const int lane = tid & 31;
    const int wave = tid >> 5;
    const int half = lane >> 4;     // 0: lanes 0-15, 1: lanes 16-31
    const int lr   = lane & 15;
    const int wm   = (wave & 3) * 32;
    const int wn   = (wave >> 2) * 64;

    const int m0 = blockIdx.y * TILE_M;
    const int n0 = blockIdx.x * TILE_N;

    // tile copy: 128 rows x 4 b128-segments, 2 segments per thread per matrix
    auto issue = [&](int buf, int k0) {
        #pragma unroll
        for (int i = 0; i < 2; ++i) {
            const int e   = tid + i * 256;
            const int row = e >> 2;
            const int seg = e & 3;
            async_copy_b128(A + (size_t)(m0 + row) * K + k0 + seg * 8,
                            &As[buf][row * LDSN + seg * 8]);
            async_copy_b128(W + (size_t)(n0 + row) * K + k0 + seg * 8,
                            &Bs[buf][row * LDSN + seg * 8]);
        }
    };

    v8f acc[2][4];
    #pragma unroll
    for (int mi = 0; mi < 2; ++mi)
        #pragma unroll
        for (int ni = 0; ni < 4; ++ni)
            #pragma unroll
            for (int r = 0; r < 8; ++r)
                acc[mi][ni][r] = 0.0f;

    const int ksteps = K / TILE_K;
    issue(0, 0);                       // prologue: fill buffer 0

    for (int kt = 0; kt < ksteps; ++kt) {
        const int buf = kt & 1;
        wait_async0();                 // my copies for `buf` complete
        __syncthreads();               // everyone's copies done; prev reads drained
        if (kt + 1 < ksteps)
            issue(buf ^ 1, (kt + 1) * TILE_K);   // overlaps with WMMA below

        // A fragment (ISA 7.12.2): lanes 0-15 K=0..7 (V0-3), 16..23 (V4-7);
        //                          lanes 16-31 K=8..15, 24..31
        Frag afrag[2];
        #pragma unroll
        for (int mi = 0; mi < 2; ++mi) {
            const unsigned short* p = &As[buf][(wm + mi * 16 + lr) * LDSN];
            afrag[mi].u[0] = *(const uint4*)(p + half * 8);
            afrag[mi].u[1] = *(const uint4*)(p + 16 + half * 8);
        }
        // B fragment: lanes 0-15 K=0..7, 16..23; lanes 16-31 K=8..15, 24..31, N=lane%16
        Frag bfrag[4];
        #pragma unroll
        for (int ni = 0; ni < 4; ++ni) {
            const unsigned short* p = &Bs[buf][(wn + ni * 16 + lr) * LDSN];
            bfrag[ni].u[0] = *(const uint4*)(p + half * 8);
            bfrag[ni].u[1] = *(const uint4*)(p + 16 + half * 8);
        }

        #pragma unroll
        for (int mi = 0; mi < 2; ++mi)
            #pragma unroll
            for (int ni = 0; ni < 4; ++ni)
                acc[mi][ni] = __builtin_amdgcn_wmma_f32_16x16x32_bf16(
                    false, afrag[mi].v, false, bfrag[ni].v,
                    (short)0, acc[mi][ni], false, false);
    }

    // C/D layout: VGPR r -> row r (lanes 0-15) / 8+r (lanes 16-31), col = lane%16
    #pragma unroll
    for (int ni = 0; ni < 4; ++ni) {
        const int col = n0 + wn + ni * 16 + lr;
        const float bv = bias[col];
        #pragma unroll
        for (int mi = 0; mi < 2; ++mi) {
            const int rbase = m0 + wm + mi * 16 + half * 8;
            #pragma unroll
            for (int r = 0; r < 8; ++r)
                C[(size_t)(rbase + r) * N + col] = acc[mi][ni][r] + bv;
        }
    }
}

// ---------------------------------------------------------------------------
// Per-token attention over the heads axis. qkv row (fp32, interleaved):
// flat = h*192 + dk*3 + {0:q,1:k,2:v}. One wave32 per token, 4 tokens/block.
// Context written as bf16 (feeds the bf16 output GEMM directly).
// ---------------------------------------------------------------------------
__global__ __launch_bounds__(128)
void attn_heads_kernel(const float* __restrict__ qkv,        // (BS, 3072) fp32
                       const int* __restrict__ mask,         // (16,16)
                       unsigned short* __restrict__ ctx)     // (BS, 1024) bf16
{
    __shared__ float s_row[4][3072];
    __shared__ float s_attn[4][16][17];

    const int tid  = threadIdx.x;
    const int lane = tid & 31;
    const int w    = tid >> 5;
    const size_t tok = (size_t)blockIdx.x * 4 + w;
    const float4* row = (const float4*)(qkv + tok * 3072);

    for (int i = lane; i < 768; i += 32)
        *(float4*)&s_row[w][i * 4] = row[i];
    __syncthreads();

    #pragma unroll
    for (int j = 0; j < 8; ++j) {
        const int p = lane * 8 + j;
        const int h = p >> 4, t = p & 15;
        float s = 0.0f;
        #pragma unroll 8
        for (int d = 0; d < 64; ++d)
            s = fmaf(s_row[w][h * 192 + d * 3], s_row[w][t * 192 + d * 3 + 1], s);
        s *= 0.125f;
        if (mask[h * 16 + t] == 0) s = -__builtin_inff();
        s_attn[w][h][t] = s;
    }
    __syncthreads();

    if (lane < 16) {
        const int h = lane;
        float mx = -__builtin_inff();
        #pragma unroll
        for (int t = 0; t < 16; ++t) mx = fmaxf(mx, s_attn[w][h][t]);
        float e[16], sum = 0.0f;
        #pragma unroll
        for (int t = 0; t < 16; ++t) { e[t] = __expf(s_attn[w][h][t] - mx); sum += e[t]; }
        const float inv = 1.0f / sum;
        #pragma unroll
        for (int t = 0; t < 16; ++t) s_attn[w][h][t] = e[t] * inv;
    }
    __syncthreads();

    #pragma unroll
    for (int i = 0; i < 32; ++i) {
        const int o = i * 32 + lane;          // o = h*64 + dk
        const int h = o >> 6, d = o & 63;
        float c = 0.0f;
        #pragma unroll
        for (int t = 0; t < 16; ++t)
            c = fmaf(s_attn[w][h][t], s_row[w][t * 192 + d * 3 + 2], c);
        ctx[tok * 1024 + o] = f2bf(c);
    }
}

extern "C" void kernel_launch(void* const* d_in, const int* in_sizes, int n_in,
                              void* d_out, int out_size, void* d_ws, size_t ws_size,
                              hipStream_t stream) {
    const float* x    = (const float*)d_in[0];   // (4,4096,1024)
    const float* Wqkv = (const float*)d_in[1];   // (3072,1024)
    const float* bqkv = (const float*)d_in[2];   // (3072,)
    const float* Wout = (const float*)d_in[3];   // (1024,1024)
    const float* bout = (const float*)d_in[4];   // (1024,)
    const int*   mask = (const int*)d_in[5];     // (1,1,16,16)
    float* out = (float*)d_out;

    const int Mrows = 4 * 4096;                  // 16384 tokens

    // workspace layout
    float*          qkv  = (float*)d_ws;                                   // 192 MB fp32
    unsigned short* ctxb = (unsigned short*)(qkv + (size_t)Mrows * 3072);  //  32 MB bf16
    unsigned short* xb   = ctxb + (size_t)Mrows * 1024;                    //  32 MB bf16
    unsigned short* wqb  = xb   + (size_t)Mrows * 1024;                    //   6 MB bf16
    unsigned short* wob  = wqb  + (size_t)3072  * 1024;                    //   2 MB bf16

    // 1) one-pass conversions to bf16
    const long nx = (long)Mrows * 1024 / 4;
    const long nq = (long)3072 * 1024 / 4;
    const long no = (long)1024 * 1024 / 4;
    convert_f32_bf16<<<dim3((unsigned)((nx + 255) / 256)), dim3(256), 0, stream>>>(
        x, (unsigned*)xb, nx);
    convert_f32_bf16<<<dim3((unsigned)((nq + 255) / 256)), dim3(256), 0, stream>>>(
        Wqkv, (unsigned*)wqb, nq);
    convert_f32_bf16<<<dim3((unsigned)((no + 255) / 256)), dim3(256), 0, stream>>>(
        Wout, (unsigned*)wob, no);

    // 2) QKV GEMM (bf16 WMMA, fp32 accum/out)
    gemm_bf16_wmma<<<dim3(3072 / TILE_N, Mrows / TILE_M), dim3(256), 0, stream>>>(
        xb, wqb, bqkv, qkv, Mrows, 3072, 1024);

    // 3) per-token heads-attention (fp32), context out in bf16
    attn_heads_kernel<<<dim3(Mrows / 4), dim3(128), 0, stream>>>(qkv, mask, ctxb);

    // 4) output GEMM
    gemm_bf16_wmma<<<dim3(1024 / TILE_N, Mrows / TILE_M), dim3(256), 0, stream>>>(
        ctxb, wob, bout, out, Mrows, 1024, 1024);
}
